// PatchDeduplicator_85796266705274
// MI455X (gfx1250) — 8-bit matrix operand order repaired and checked by reading, compiled, not run on hardware
//
#include <hip/hip_runtime.h>

#define PATCH 16
#define S_    16
#define C_    3
#define H_    720
#define W_    1280
#define NH    45
#define NW    80
#define P_    (NH * NW)
#define FP    768
#define ROW   770
#define NPC   (S_ * C_ * P_)

typedef __attribute__((ext_vector_type(8))) int   v8i;
typedef __attribute__((ext_vector_type(4))) int   v4i;
typedef __attribute__((ext_vector_type(4))) float v4f;
typedef __attribute__((ext_vector_type(2))) float v2f;

__global__ void __launch_bounds__(256)
patchdedup_fused(const float* __restrict__ frames,
                 float* __restrict__ out,
                 float* __restrict__ down)
{
    __shared__ unsigned char lds[8 * 4096];

    const int lane = threadIdx.x & 31;
    const int wave = threadIdx.x >> 5;
    const int gwave = blockIdx.x * 8 + wave;
    const int pcbase = gwave * 16;
    unsigned char* L = lds + wave * 4096;

    #pragma unroll
    for (int i = 0; i < 8; ++i) {
        const int r  = lane + 32 * i;
        const int n  = r >> 4;
        const int pi = r & 15;
        const int pc = pcbase + n;
        const int p  = pc % P_;
        const int sc = pc / P_;
        const int c  = sc % C_;
        const int s  = sc / C_;
        const int ph = p / NW, pw = p % NW;

        const long long ibase =
            (((long long)sc * H_) + (long long)(ph * PATCH + pi)) * W_
            + pw * PATCH;
        const v4f* src = (const v4f*)(frames + ibase);

        const long long obase =
            ((long long)s * P_ + p) * ROW + c * 256 + pi * 16;
        v2f* dst = (v2f*)(out + obase);

        #pragma unroll
        for (int q = 0; q < 4; ++q) {
            v4f v = __builtin_nontemporal_load(src + q);
            unsigned char b0 = (unsigned char)(v.x * 255.0f);
            unsigned char b1 = (unsigned char)(v.y * 255.0f);
            unsigned char b2 = (unsigned char)(v.z * 255.0f);
            unsigned char b3 = (unsigned char)(v.w * 255.0f);
            const int pj = q * 4;
            L[(pi * 16 + pj + 0) * 16 + n] = b0;
            L[(pi * 16 + pj + 1) * 16 + n] = b1;
            L[(pi * 16 + pj + 2) * 16 + n] = b2;
            L[(pi * 16 + pj + 3) * 16 + n] = b3;
            v2f lo2 = {(float)b0, (float)b1};
            v2f hi2 = {(float)b2, (float)b3};
            __builtin_nontemporal_store(lo2, dst + 2 * q + 0);
            __builtin_nontemporal_store(hi2, dst + 2 * q + 1);
        }
    }

    __builtin_amdgcn_wave_barrier();
    asm volatile("s_wait_dscnt 0x0" ::: "memory");
    __builtin_amdgcn_wave_barrier();

    v8i a;
    #pragma unroll
    for (int i = 0; i < 8; ++i) a[i] = 0x01010101;
    v8i acc = {};

    const v4i* Lv = (const v4i*)L;
    #pragma unroll
    for (int j = 0; j < 4; ++j) {
        const int t0 = 64 * j + 32 * (lane >> 4);
        const int nn = lane & 15;
        v8i b;
        #pragma unroll
        for (int w = 0; w < 8; ++w) {
            unsigned int u = 0;
            #pragma unroll
            for (int e = 0; e < 4; ++e)
                u |= (unsigned int)L[(t0 + 4 * w + e) * 16 + nn] << (8 * e);
            b[w] = (int)u;
        }
        acc = __builtin_amdgcn_wmma_i32_16x16x64_iu8(
                   false, a,  false, b, acc,
                   false,  false);
    }

    if (lane < 16) {
        down[pcbase + lane] = (float)acc[0] * (1.0f / 256.0f);
    }
}

__global__ void __launch_bounds__(256)
patchdedup_mask(const float* __restrict__ down, float* __restrict__ out)
{
    const int tid = blockIdx.x * blockDim.x + threadIdx.x;
    if (tid >= S_ * P_) return;
    const int p = tid % P_;
    const int s = tid / P_;
    const int x = p % NW;
    const int y = p / NW;

    const long long base = ((long long)s * P_ + p) * ROW;
    out[base + 768] = (float)x;
    out[base + 769] = (float)y;

    float m;
    if (s == 0) {
        m = 1.0f;
    } else {
        float d = 0.0f;
        #pragma unroll
        for (int c = 0; c < C_; ++c) {
            d += fabsf(down[(s * C_ + c) * P_ + p] -
                       down[((s - 1) * C_ + c) * P_ + p]);
        }
        m = (d * (1.0f / 3.0f) > 2.0f) ? 1.0f : 0.0f;
    }
    out[(long long)S_ * P_ * ROW + tid] = m;
}

extern "C" void kernel_launch(void* const* d_in, const int* in_sizes, int n_in,
                              void* d_out, int out_size, void* d_ws, size_t ws_size,
                              hipStream_t stream)
{
    const float* frames = (const float*)d_in[0];
    float* out  = (float*)d_out;
    float* down = (float*)d_ws;

    patchdedup_fused<<<dim3(1350), dim3(256), 0, stream>>>(frames, out, down);

    patchdedup_mask<<<dim3((S_ * P_ + 255) / 256), dim3(256), 0, stream>>>(down, out);
}
